// ScaledDotProductAttention_64673617543185
// MI455X (gfx1250) — matrix operand order repaired and checked by reading, not compiled, not run on hardware
//
#include <hip/hip_runtime.h>

typedef __attribute__((ext_vector_type(16))) _Float16 v16h;
typedef __attribute__((ext_vector_type(8)))  _Float16 v8h;
typedef __attribute__((ext_vector_type(2)))  _Float16 v2h;
typedef __attribute__((ext_vector_type(8)))  float    v8f;
typedef __attribute__((ext_vector_type(4)))  float    v4f;

#define N_B   32
#define SEQ   2048
#define HD    128
#define RS    (3 * HD)     // 384 floats per (batch, seq) row of qkv
#define CHUNK 64           // keys per outer iteration
#define NITER (SEQ / CHUNK)

// Block = 256 threads = 8 waves; each wave owns 16 query rows (128 q rows / block).
// Grid = N_B * (SEQ/128) = 512 blocks.
//
// Scores computed TRANSPOSED: S^T(key,q) = K_tile x Q^T via v_wmma_f32_16x16x32_f16.
// D layout: M(key)=vgpr+8*(lane>=16), N(q)=lane&15  =>
//   * per-query softmax stats are in-lane + one shfl_xor(16)
//   * exp'd probabilities land exactly in the A-operand layout for P x V.
// K/V staged through double-buffered LDS (f16, V transposed); next chunk's global
// loads are issued right after the barrier and overlap the current chunk's WMMAs.

__global__ __launch_bounds__(256)
void fa_fwd_kernel(const float* __restrict__ qkv,
                   const float* __restrict__ mask,
                   float* __restrict__ out)
{
    __shared__ __align__(32) _Float16 Klds[2][CHUNK * HD];  // [key][dim]  2 x 16 KB
    __shared__ __align__(32) _Float16 Vt[2][HD * CHUNK];    // [dim][key]  2 x 16 KB

    const int tid  = threadIdx.x;
    const int lane = tid & 31;
    const int n    = lane & 15;
    const int hi   = lane >> 4;
    const int wave = tid >> 5;

    const int blocksPerBatch = SEQ / 128;            // 16
    const int batch  = blockIdx.x / blocksPerBatch;
    const int qblk   = blockIdx.x % blocksPerBatch;
    const int q_base = qblk * 128 + wave * 16;

    const float* qkvB = qkv + (size_t)batch * SEQ * RS;

    const float LOG2E  = 1.44269504088896340736f;
    const float qscale = 0.08838834764831845f * 1.44269504088896340736f; // log2e/sqrt(HD)

    // ---- per-thread staging offsets (within one chunk), computed once ----
    int kOff[4], kSt[4];                 // K: 4 units of (key, 8 dims)
#pragma unroll
    for (int i = 0; i < 4; ++i) {
        const int u = tid + i * 256;
        kOff[i] = (u >> 4) * RS + HD + (u & 15) * 8;    // float offset in qkv chunk
        kSt[i]  = (u >> 4) * HD + (u & 15) * 8;         // half offset in Klds
    }
    int vOff[2], vKp[2], vDg[2];         // V: 2 units of (key-pair, 8 dims)
#pragma unroll
    for (int i = 0; i < 2; ++i) {
        const int u = tid + i * 256;
        vKp[i]  = u & 31;
        vDg[i]  = u >> 5;
        vOff[i] = vKp[i] * 2 * RS + 2 * HD + vDg[i] * 8;
    }

    // ---- Q fragment (B operand of S^T gemm), scaled into exp2 domain ----
    v16h bq[4];
#pragma unroll
    for (int kc = 0; kc < 4; ++kc) {
        const float* qp = qkvB + (size_t)(q_base + n) * RS + kc * 32 + hi * 16;
#pragma unroll
        for (int j = 0; j < 16; ++j)
            bq[kc][j] = (_Float16)(qp[j] * qscale);
    }

    v8f acc[8];
#pragma unroll
    for (int t = 0; t < 8; ++t)
        acc[t] = (v8f){0.f, 0.f, 0.f, 0.f, 0.f, 0.f, 0.f, 0.f};

    float m_i = -3.0e38f;
    float l_i = 0.0f;

    const float* mrow = mask + ((size_t)batch * SEQ + (size_t)(q_base + n)) * SEQ;

    // ---- staged chunk held in registers between load and LDS store ----
    v4f ka[4], kb[4], va0[2], va1[2], vb0[2], vb1[2];

    auto load_chunk = [&](int kvbase) {
#pragma unroll
        for (int i = 0; i < 4; ++i) {
            const float* p = qkvB + (size_t)kvbase * RS + kOff[i];
            ka[i] = *(const v4f*)p;
            kb[i] = *(const v4f*)(p + 4);
        }
#pragma unroll
        for (int i = 0; i < 2; ++i) {
            const float* p0 = qkvB + (size_t)kvbase * RS + vOff[i];
            const float* p1 = p0 + RS;
            va0[i] = *(const v4f*)p0;  va1[i] = *(const v4f*)(p0 + 4);
            vb0[i] = *(const v4f*)p1;  vb1[i] = *(const v4f*)(p1 + 4);
        }
    };

    auto store_chunk = [&](int buf) {
#pragma unroll
        for (int i = 0; i < 4; ++i) {
            v8h h;
#pragma unroll
            for (int d = 0; d < 4; ++d) { h[d] = (_Float16)ka[i][d]; h[d + 4] = (_Float16)kb[i][d]; }
            *(v8h*)(&Klds[buf][kSt[i]]) = h;
        }
#pragma unroll
        for (int i = 0; i < 2; ++i) {
#pragma unroll
            for (int dd = 0; dd < 8; ++dd) {
                const float x = (dd < 4) ? va0[i][dd] : va1[i][dd - 4];
                const float y = (dd < 4) ? vb0[i][dd] : vb1[i][dd - 4];
                v2h h; h[0] = (_Float16)x; h[1] = (_Float16)y;
                *(v2h*)(&Vt[buf][(vDg[i] * 8 + dd) * CHUNK + 2 * vKp[i]]) = h;
            }
        }
    };

    load_chunk(0);

    for (int it = 0; it < NITER; ++it) {
        const int kv  = it * CHUNK;
        const int buf = it & 1;

        store_chunk(buf);
        __syncthreads();                 // waits dscnt=0, then barrier signal+wait

        if (it + 1 < NITER)
            load_chunk(kv + CHUNK);      // overlap next chunk's global loads w/ WMMAs

        // ================= scores: four 16(key) x 16(q) tiles =================
        v8f s[4];
#pragma unroll
        for (int t2 = 0; t2 < 4; ++t2) {
            v8f sc = (v8f){0.f, 0.f, 0.f, 0.f, 0.f, 0.f, 0.f, 0.f};
#pragma unroll
            for (int kc = 0; kc < 4; ++kc) {
                const _Float16* kptr = &Klds[buf][(t2 * 16 + n) * HD + kc * 32 + hi * 16];
                v8h lo = *(const v8h*)kptr;
                v8h hh = *(const v8h*)(kptr + 8);
                v16h ak = __builtin_shufflevector(lo, hh, 0, 1, 2, 3, 4, 5, 6, 7,
                                                          8, 9, 10, 11, 12, 13, 14, 15);
                sc = __builtin_amdgcn_wmma_f32_16x16x32_f16(
                        false, ak, false, bq[kc], (short)0, sc, false, false);
            }
            // mask (streaming NT): fma into exp2 domain
            const float* mp = mrow + kv + t2 * 16 + hi * 8;
            v4f m0 = __builtin_nontemporal_load((const v4f*)mp);
            v4f m1 = __builtin_nontemporal_load((const v4f*)(mp + 4));
#pragma unroll
            for (int v = 0; v < 4; ++v) {
                sc[v]     = fmaf(m0[v], LOG2E, sc[v]);
                sc[v + 4] = fmaf(m1[v], LOG2E, sc[v + 4]);
            }
            s[t2] = sc;
        }

        // ================= online softmax (exp2 domain, per q = lane&15) =================
        float mx = s[0][0];
#pragma unroll
        for (int t2 = 0; t2 < 4; ++t2)
#pragma unroll
            for (int v = 0; v < 8; ++v) mx = fmaxf(mx, s[t2][v]);
        mx = fmaxf(mx, __shfl_xor(mx, 16));

        const float m_new = fmaxf(m_i, mx);
        const float alpha = __builtin_amdgcn_exp2f(m_i - m_new);

        float rs = 0.f;
#pragma unroll
        for (int t2 = 0; t2 < 4; ++t2)
#pragma unroll
            for (int v = 0; v < 8; ++v) {
                s[t2][v] = __builtin_amdgcn_exp2f(s[t2][v] - m_new);
                rs += s[t2][v];
            }
        rs += __shfl_xor(rs, 16);
        l_i = l_i * alpha + rs;
        m_i = m_new;

        // probabilities already in A layout: one fragment per 32-key sub-chunk
        v16h ap[2];
#pragma unroll
        for (int g = 0; g < 2; ++g)
#pragma unroll
            for (int j = 0; j < 8; ++j) {
                ap[g][j]     = (_Float16)s[2 * g][j];
                ap[g][j + 8] = (_Float16)s[2 * g + 1][j];
            }

        // rescale accumulator (vectorized -> v_pk_mul_f32): acc row q = v + 8*hi
        v8f avv;
#pragma unroll
        for (int v = 0; v < 8; ++v) avv[v] = __shfl(alpha, v + 8 * hi);
#pragma unroll
        for (int t = 0; t < 8; ++t) acc[t] = acc[t] * avv;

        // ================= P x V : 2 key sub-chunks x 8 dim-tiles =================
#pragma unroll
        for (int g = 0; g < 2; ++g)
#pragma unroll
            for (int t = 0; t < 8; ++t) {
                const _Float16* vptr = &Vt[buf][(t * 16 + n) * CHUNK + g * 32 + hi * 8];
                v16h bv = __builtin_shufflevector(*(const v8h*)vptr, *(const v8h*)(vptr + 16), 0, 1, 2, 3, 4, 5, 6, 7, 8, 9, 10, 11, 12, 13, 14, 15);
                acc[t] = __builtin_amdgcn_wmma_f32_16x16x32_f16(
                            false, ap[g], false, bv, (short)0, acc[t], false, false);
            }
    }

    // ================= normalize + store =================
    v8f invv;
#pragma unroll
    for (int v = 0; v < 8; ++v) invv[v] = 1.0f / __shfl(l_i, v + 8 * hi);

#pragma unroll
    for (int t = 0; t < 8; ++t) {
        const v8f r = acc[t] * invv;
#pragma unroll
        for (int v = 0; v < 8; ++v) {
            const int qrow = q_base + v + 8 * hi;
            out[((size_t)batch * SEQ + (size_t)qrow) * HD + t * 16 + n] = r[v];
        }
    }
}

extern "C" void kernel_launch(void* const* d_in, const int* in_sizes, int n_in,
                              void* d_out, int out_size, void* d_ws, size_t ws_size,
                              hipStream_t stream) {
    (void)in_sizes; (void)n_in; (void)out_size; (void)d_ws; (void)ws_size;
    const float* qkv  = (const float*)d_in[0];
    const float* mask = (const float*)d_in[1];
    float* out        = (float*)d_out;

    dim3 grid(N_B * (SEQ / 128));   // 512 blocks
    dim3 block(256);                // 8 waves of 32
    fa_fwd_kernel<<<grid, block, 0, stream>>>(qkv, mask, out);
}
